// CrossAttention_40845138985031
// MI455X (gfx1250) — matrix operand order repaired and checked by reading, not compiled, not run on hardware
//
#include <hip/hip_runtime.h>

// ---------------------------------------------------------------------------
// MI455X (gfx1250, wave32) flash-attention for the cross-attention reference.
//   ~17.2 GFLOP attention (N=4096, d=64, B=4), bf16 WMMA w/ fp32 accumulate.
//   v4: double-buffered K/V^T tiles staged with GLOBAL_LOAD_ASYNC_TO_LDS_B128
//       (ASYNCcnt) so the DMA of tile k+1 overlaps the WMMAs of tile k;
//       exp2-domain softmax (log2e folded into K projection, raw v_exp_f32).
// ---------------------------------------------------------------------------

typedef __attribute__((ext_vector_type(16))) __bf16 v16bf;
typedef __attribute__((ext_vector_type(8)))  __bf16 bf8;     // 16 bytes
typedef __attribute__((ext_vector_type(8)))  float  v8f;
typedef __attribute__((ext_vector_type(4)))  int    v4i;

#define AS1 __attribute__((address_space(1)))
#define AS3 __attribute__((address_space(3)))

#define B_    4
#define C_    64
#define HH_   64
#define WW_   64
#define NTOK  (C_ * HH_)       // 4096 tokens per batch
#define PB    (NTOK * WW_)     // 262144 elements per batch per tensor
#define NT64  (NTOK / 64)      // 64 key tiles

#define LOG2E 1.4426950408889634f

// --- raw 2^x (v_exp_f32); args are always <= 0 here, flush-to-zero is fine --
__device__ __forceinline__ float fast_exp2(float x) {
#if defined(__gfx1250__) && __has_builtin(__builtin_amdgcn_exp2f)
  return __builtin_amdgcn_exp2f(x);
#else
  return exp2f(x);
#endif
}

// --- async global -> LDS, 16 bytes per lane ---------------------------------
__device__ __forceinline__ void async_cp16(__bf16* l, const __bf16* g) {
#if defined(__gfx1250__)
#if __has_builtin(__builtin_amdgcn_global_load_async_to_lds_b128)
  // signature (from compiler diagnostic): (v4i AS1*, v4i AS3*, Imm offset, Imm cpol)
  __builtin_amdgcn_global_load_async_to_lds_b128(
      (AS1 v4i*)g, (AS3 v4i*)l, 0, 0);
#else
  unsigned int lofs = (unsigned int)(size_t)(AS3 char*)(void*)l;
  unsigned long long ga = (unsigned long long)(size_t)g;
  asm volatile("global_load_async_to_lds_b128 %0, %1, off"
               :: "v"(lofs), "v"(ga) : "memory");
#endif
#else
  *(bf8*)l = *(const bf8*)g;   // host-pass / non-gfx1250 fallback
#endif
}

__device__ __forceinline__ void wait_async() {
#if defined(__gfx1250__)
#if __has_builtin(__builtin_amdgcn_s_wait_asynccnt)
  __builtin_amdgcn_s_wait_asynccnt(0);
#else
  asm volatile("s_wait_asynccnt 0x0" ::: "memory");
#endif
#endif
}

// A-operand fragment (16-bit 16x32): per lane 16 contiguous elems (K=16h..16h+15)
// (lane half already folded into p by caller: p = row*ld + 16*h).
__device__ __forceinline__ v16bf mk_afrag(const __bf16* p) {
  bf8 lo = *(const bf8*)(p);
  bf8 hi = *(const bf8*)(p + 8);
  return __builtin_shufflevector(lo, hi, 0,1,2,3,4,5,6,7,8,9,10,11,12,13,14,15);
}

// B-operand fragment (16-bit 32x16): per lane 16 contiguous elems (K=16h..16h+15).
__device__ __forceinline__ v16bf mk_bfrag(const __bf16* p) {
  bf8 lo = *(const bf8*)(p);
  bf8 hi = *(const bf8*)(p + 8);
  return __builtin_shufflevector(lo, hi, 0,1,2,3,4,5,6,7,8,9,10,11,12,13,14,15);
}

// ---------------------------------------------------------------------------
// Kernel 1: fused Q/K/V 1x1 conv. One block per (b,h). Emits bf16:
//   Q[b, n, w], K[b, n, w] row-major (K pre-scaled by log2e), VT[b, w, n].
// ---------------------------------------------------------------------------
__global__ __launch_bounds__(256) void qkv_conv_kernel(
    const float* __restrict__ v1, const float* __restrict__ v2,
    const float* __restrict__ wq, const float* __restrict__ bq,
    const float* __restrict__ wk, const float* __restrict__ bk,
    const float* __restrict__ wv, const float* __restrict__ bv,
    void* __restrict__ ws)
{
  __bf16* Q  = (__bf16*)ws;
  __bf16* K  = Q + (size_t)B_ * PB;
  __bf16* VT = K + (size_t)B_ * PB;

  __shared__ float X1[C_][WW_];
  __shared__ float X2[C_][WW_];

  const int t = threadIdx.x;
  const int b = blockIdx.x >> 6;
  const int h = blockIdx.x & 63;

  const float* x1 = v1 + (size_t)b * PB + h * WW_;  // [c][w] at c*4096 + w
  const float* x2 = v2 + (size_t)b * PB + h * WW_;

  {
    const int c  = t >> 2;
    const int w0 = (t & 3) * 16;
#pragma unroll
    for (int j = 0; j < 16; j += 4) {
      *(float4*)&X1[c][w0 + j] = *(const float4*)(x1 + (size_t)c * HH_ * WW_ + w0 + j);
      *(float4*)&X2[c][w0 + j] = *(const float4*)(x2 + (size_t)c * HH_ * WW_ + w0 + j);
    }
  }
  __syncthreads();

  const int o  = t >> 2;            // output channel
  const int w0 = (t & 3) * 16;

  float aq[16], ak[16], av[16];
#pragma unroll
  for (int j = 0; j < 16; ++j) {
    aq[j] = bq[o];
    ak[j] = bk[o] * LOG2E;          // exp2-domain softmax
    av[j] = bv[o];
  }

  for (int c = 0; c < C_; ++c) {
    const float wqv = wq[o * C_ + c];
    const float wkv = wk[o * C_ + c] * LOG2E;
    const float wvv = wv[o * C_ + c];
#pragma unroll
    for (int j = 0; j < 16; ++j) {
      const float x1v = X1[c][w0 + j];
      const float x2v = X2[c][w0 + j];
      aq[j] = fmaf(wqv, x1v, aq[j]);
      ak[j] = fmaf(wkv, x2v, ak[j]);
      av[j] = fmaf(wvv, x2v, av[j]);
    }
  }

  const int n = o * HH_ + h;        // token index (c-major, matches reshape)
  __bf16* qrow = Q + (size_t)b * PB + (size_t)n * WW_ + w0;
  __bf16* krow = K + (size_t)b * PB + (size_t)n * WW_ + w0;
  __bf16* vcol = VT + (size_t)b * PB + n;     // VT row stride = NTOK
#pragma unroll
  for (int j = 0; j < 16; ++j) {
    qrow[j] = (__bf16)aq[j];
    krow[j] = (__bf16)ak[j];
    vcol[(size_t)(w0 + j) * NTOK] = (__bf16)av[j];
  }
}

// ---------------------------------------------------------------------------
// Kernel 2: flash attention. 8 waves/block, 16 queries/wave, 128 queries/block.
// Double-buffered 64-key tiles; async global->LDS overlaps WMMA compute.
// ---------------------------------------------------------------------------
__global__ __launch_bounds__(256) void flash_attn_kernel(
    const void* __restrict__ ws, float* __restrict__ out)
{
  const __bf16* Q  = (const __bf16*)ws;
  const __bf16* K  = Q + (size_t)B_ * PB;
  const __bf16* VT = K + (size_t)B_ * PB;

  __shared__ __bf16 Ks[2][64 * 72];   // 64 keys x 64 w   (padded stride)
  __shared__ __bf16 Vs[2][64 * 72];   // 64 w    x 64 keys (padded stride)

  const int t    = threadIdx.x;
  const int lane = t & 31;
  const int wid  = t >> 5;
  const int hh   = lane >> 4;      // lane half
  const int ln   = lane & 15;

  const int b      = blockIdx.x >> 5;
  const int q_base = (blockIdx.x & 31) * 128 + wid * 16;

  const __bf16* Qb = Q  + (size_t)b * PB;
  const __bf16* Kb = K  + (size_t)b * PB;
  const __bf16* Vb = VT + (size_t)b * PB;

  const int r  = t >> 2;           // staging row for this thread
  const int cc = (t & 3) * 16;     // staging column chunk

  auto stage = [&](int buf, int key0) {
    __bf16*       kd = &Ks[buf][r * 72 + cc];
    __bf16*       vd = &Vs[buf][r * 72 + cc];
    const __bf16* kg = Kb + (size_t)(key0 + r) * WW_ + cc;
    const __bf16* vg = Vb + (size_t)r * NTOK + key0 + cc;
    async_cp16(kd,     kg);
    async_cp16(kd + 8, kg + 8);
    async_cp16(vd,     vg);
    async_cp16(vd + 8, vg + 8);
  };

  // Q as B-operand of S^T = K x Q^T: lane = query column, 16 contiguous w.
  v16bf qf[2];
  {
    const __bf16* qrow = Qb + (size_t)(q_base + ln) * WW_ + 16 * hh;
    qf[0] = mk_bfrag(qrow);
    qf[1] = mk_bfrag(qrow + 32);
  }

  float m_run = -3.0e38f;
  float l_run = 0.0f;
  v8f acc[4];                      // O^T: 4 w-tiles of 16x16 fp32
#pragma unroll
  for (int wt = 0; wt < 4; ++wt) acc[wt] = 0.0f;

  // Prologue: stage tile 0 into buffer 0.
  stage(0, 0);
  wait_async();
  __syncthreads();

  for (int kt = 0; kt < NT64; ++kt) {
    const int cur = kt & 1;

    // Kick off DMA for the next tile while we compute on this one.
    if (kt + 1 < NT64) stage(cur ^ 1, (kt + 1) * 64);
    if (kt + 2 < NT64) {           // -> global_prefetch_b8, warm L2 for kt+2
      __builtin_prefetch(Kb + (size_t)((kt + 2) * 64 + r) * WW_ + cc, 0, 1);
      __builtin_prefetch(Vb + (size_t)r * NTOK + (kt + 2) * 64 + cc, 0, 1);
    }

    const __bf16* KsC = &Ks[cur][0];
    const __bf16* VsC = &Vs[cur][0];

    // S^T tiles: s[mt] = K[key0+16mt : +16, :] . Q^T   (K-dim = 64, 2 WMMAs)
    v8f s[4];
#pragma unroll
    for (int mt = 0; mt < 4; ++mt) {
      const __bf16* arow = KsC + (mt * 16 + ln) * 72 + 16 * hh;
      const v16bf a0 = mk_afrag(arow);
      const v16bf a1 = mk_afrag(arow + 32);
      v8f z = 0.0f;
      s[mt] = __builtin_amdgcn_wmma_f32_16x16x32_bf16(false, a0, false, qf[0],
                                                      (short)0, z, false, false);
      s[mt] = __builtin_amdgcn_wmma_f32_16x16x32_bf16(false, a1, false, qf[1],
                                                      (short)0, s[mt], false, false);
    }

    // Online softmax over keys (M dim of S^T): 32 in-lane values + xor16.
    float tmax = -3.0e38f;
#pragma unroll
    for (int mt = 0; mt < 4; ++mt)
#pragma unroll
      for (int v = 0; v < 8; ++v) tmax = fmaxf(tmax, s[mt][v]);
    tmax = fmaxf(tmax, __shfl_xor(tmax, 16, 32));

    const float m_new = fmaxf(m_run, tmax);
    const float alpha = fast_exp2(m_run - m_new);  // logits already in log2 domain
    float rsum = 0.0f;
#pragma unroll
    for (int mt = 0; mt < 4; ++mt)
#pragma unroll
      for (int v = 0; v < 8; ++v) {
        const float p = fast_exp2(s[mt][v] - m_new);
        s[mt][v] = p;
        rsum += p;
      }
    rsum += __shfl_xor(rsum, 16, 32);
    l_run = l_run * alpha + rsum;
    m_run = m_new;
#pragma unroll
    for (int wt = 0; wt < 4; ++wt)
#pragma unroll
      for (int v = 0; v < 8; ++v) acc[wt][v] *= alpha;

    // P^T as B-operand: lane L needs tile (2kc+hh); one xor-16 exchange.
    v16bf bp[2];
#pragma unroll
    for (int kc = 0; kc < 2; ++kc) {
#pragma unroll
      for (int v = 0; v < 8; ++v) {
        const float keep = hh ? s[2 * kc + 1][v] : s[2 * kc][v];
        const float send = hh ? s[2 * kc][v]     : s[2 * kc + 1][v];
        const float rec  = __shfl_xor(send, 16, 32);
        const float loe  = hh ? rec  : keep;     // element v     (m' = v)
        const float hie  = hh ? keep : rec;      // element v+8   (m' = v+8)
        bp[kc][v]     = (__bf16)loe;
        bp[kc][v + 8] = (__bf16)hie;
      }
    }

    // O^T += V^T . P^T : 4 w-tiles, K-dim = 64 keys (2 WMMAs each)
#pragma unroll
    for (int wt = 0; wt < 4; ++wt) {
      const __bf16* vrow = VsC + (wt * 16 + ln) * 72 + 16 * hh;
      const v16bf a0 = mk_afrag(vrow);
      const v16bf a1 = mk_afrag(vrow + 32);
      acc[wt] = __builtin_amdgcn_wmma_f32_16x16x32_bf16(false, a0, false, bp[0],
                                                        (short)0, acc[wt], false, false);
      acc[wt] = __builtin_amdgcn_wmma_f32_16x16x32_bf16(false, a1, false, bp[1],
                                                        (short)0, acc[wt], false, false);
    }

    // Next tile's async loads had the whole compute phase to land.
    wait_async();
    __syncthreads();
  }

  // Epilogue: out[n, w] = O^T[w, n] / l.  Per lane: fixed n, contiguous w runs.
  const float invl = 1.0f / l_run;
  const int   n    = q_base + ln;
  float* orow = out + (size_t)b * PB + (size_t)n * WW_;
#pragma unroll
  for (int wt = 0; wt < 4; ++wt) {
    const int w0 = wt * 16 + 8 * hh;
    float4 lo4 = make_float4(acc[wt][0] * invl, acc[wt][1] * invl,
                             acc[wt][2] * invl, acc[wt][3] * invl);
    float4 hi4 = make_float4(acc[wt][4] * invl, acc[wt][5] * invl,
                             acc[wt][6] * invl, acc[wt][7] * invl);
    *(float4*)(orow + w0)     = lo4;
    *(float4*)(orow + w0 + 4) = hi4;
  }
}

// ---------------------------------------------------------------------------
extern "C" void kernel_launch(void* const* d_in, const int* in_sizes, int n_in,
                              void* d_out, int out_size, void* d_ws, size_t ws_size,
                              hipStream_t stream) {
  const float* v1 = (const float*)d_in[0];
  const float* v2 = (const float*)d_in[1];
  const float* wq = (const float*)d_in[2];
  const float* bq = (const float*)d_in[3];
  const float* wk = (const float*)d_in[4];
  const float* bk = (const float*)d_in[5];
  const float* wv = (const float*)d_in[6];
  const float* bv = (const float*)d_in[7];
  float* out = (float*)d_out;

  // Workspace: Q (2MB) | K (2MB) | VT (2MB), all bf16.
  qkv_conv_kernel<<<B_ * HH_, 256, 0, stream>>>(v1, v2, wq, bq, wk, bk, wv, bv, d_ws);
  flash_attn_kernel<<<B_ * (NTOK / 128), 256, 0, stream>>>(d_ws, out);
}
